// GPTBigCodeAttentionFused_32676111188297
// MI455X (gfx1250) — matrix operand order repaired and checked by reading, not compiled, not run on hardware
//
#include <hip/hip_runtime.h>

// ---------------------------------------------------------------------------
// GPT-BigCode MQA fused attention for gfx1250 (MI455X), bf16 WMMA pipeline.
//   k1: qkv = x @ w_attn + b_attn          (f32 in -> bf16 out, WMMA bf16)
//   k2: flash MQA attention                (bf16, WMMA bf16, online softmax,
//                                           K tiles staged via TDM tensor DMA)
//   k3: out = attn @ w_proj + b_proj       (bf16 in -> f32 out, WMMA bf16)
// ---------------------------------------------------------------------------

typedef __bf16 bf16_t;
typedef __attribute__((ext_vector_type(16))) __bf16 v16bf;
typedef __attribute__((ext_vector_type(8)))  __bf16 v8bf;
typedef __attribute__((ext_vector_type(4)))  __bf16 v4bf;
typedef __attribute__((ext_vector_type(8)))  float  v8f;
typedef __attribute__((ext_vector_type(4)))  float  v4f;
typedef __attribute__((ext_vector_type(4)))  unsigned int v4u;
typedef __attribute__((ext_vector_type(8)))  int    v8i;
typedef __attribute__((ext_vector_type(4)))  int    v4i;

#define BATCH   2
#define SEQ     2048
#define DMODEL  2048
#define NHEAD   16
#define HDIM    128
#define QKVW    2304      // D + 2*HD
#define KCOL    2048      // k offset within qkv row
#define VCOL    2176      // v offset within qkv row
#define KSTR    136       // Ksm row stride (elements) = 128 + 8 pad

__device__ __forceinline__ bf16_t f2bf(float f) {
  unsigned u = __builtin_bit_cast(unsigned, f);
  unsigned r = u + 0x7fffu + ((u >> 16) & 1u);   // round-to-nearest-even
  unsigned short s = (unsigned short)(r >> 16);
  return __builtin_bit_cast(bf16_t, s);
}

__device__ __forceinline__ v8f wmma_bf16(v16bf a, v16bf b, v8f c) {
  return __builtin_amdgcn_wmma_f32_16x16x32_bf16(false, a, false, b,
                                                 (short)0, c, false, false);
}

// A-matrix 16x32 bf16 fragment (ISA 7.12.2):
// lane l: row M = l%16; elems i=0..7 -> K = half*8 + i ; i=8..15 -> K = 16 + half*8 + (i-8)
__device__ __forceinline__ v16bf frag_a(const bf16_t* __restrict__ base,
                                        int stride, int lane) {
  const int row = lane & 15;
  const int kh  = (lane >> 4) << 3;
  const bf16_t* p = base + row * stride + kh;
  v8bf lo = *(const v8bf*)(p);
  v8bf hi = *(const v8bf*)(p + 16);
  return __builtin_shufflevector(lo, hi, 0,1,2,3,4,5,6,7,8,9,10,11,12,13,14,15);
}

// B-matrix 32x16 bf16 fragment, sourced from LDS stored as [n][k] (k contiguous):
// lane l: col N = l%16; elems i=0..7 -> K = half*8 + i ; i=8..15 -> K = 16 + half*8 + (i-8)
__device__ __forceinline__ v16bf frag_b(const bf16_t* __restrict__ base,
                                        int stride, int lane) {
  const int row = lane & 15;
  const int kh  = (lane >> 4) << 3;
  const bf16_t* p = base + row * stride + kh;
  v8bf lo = *(const v8bf*)(p);
  v8bf hi = *(const v8bf*)(p + 16);
  return __builtin_shufflevector(lo, hi, 0,1,2,3,4,5,6,7,8,9,10,11,12,13,14,15);
}

// ---------------------------------------------------------------------------
// TDM: DMA a 2D bf16 tile [32 x 128] (row stride QKVW elems) from global
// memory into LDS, padding each 256B row with 16B so the LDS stride is
// KSTR(136) elements. Issued once per wave (EXEC-independent); completion via
// s_wait_tensorcnt. D# layout per CDNA5 ISA.
// This toolchain exposes the 6-arg builtin (extra zero-filled group).
// ---------------------------------------------------------------------------
__device__ __forceinline__ void tdm_load_k_tile(const bf16_t* gsrc,
                                                unsigned lds_byte_off) {
  const unsigned long long ga = (unsigned long long)(const void*)gsrc;
  v4u g0;
  g0[0] = 1u;                                  // count=1, user descriptor
  g0[1] = lds_byte_off;                        // lds_addr
  g0[2] = (unsigned)(ga & 0xffffffffu);        // global_addr[31:0]
  g0[3] = (unsigned)((ga >> 32) & 0x01ffffffu) // global_addr[56:32]
        | (2u << 30);                          // type = 2 ("image")
  v8i g1;
  g1[0] = (1 << 16)                            // data_size: 2 bytes
        | (1 << 20)                            // pad_enable
        | (5 << 22)                            // pad_interval: 64 DWORDs (256B row)
        | (3 << 25);                           // pad_amount: 4 DWORDs (16B)
  g1[1] = (int)(128u << 16);                   // tensor_dim0[15:0] = 128
  g1[2] = (int)(32u << 16);                    // tensor_dim0 hi=0 | tensor_dim1 lo = 32
  g1[3] = (int)(128u << 16);                   // tensor_dim1 hi=0 | tile_dim0 = 128
  g1[4] = 32;                                  // tile_dim1 = 32, tile_dim2 = 0
  g1[5] = QKVW;                                // tensor_dim0_stride[31:0]
  g1[6] = 0;                                   // stride hi / tensor_dim1_stride lo
  g1[7] = 0;
  v4i gz4;
  gz4[0] = 0; gz4[1] = 0; gz4[2] = 0; gz4[3] = 0;  // groups 2/3 unused (2D tile)
  v8i gz8;
  #pragma unroll
  for (int i = 0; i < 8; ++i) gz8[i] = 0;
  __builtin_amdgcn_tensor_load_to_lds(g0, g1, gz4, gz4, gz8, 0);
}

// ---------------------------------------------------------------------------
// Generic GEMM: C[M,N] = A[M,K] * W[K,N] + bias[N]
// Block tile 128x128, K-step 32. 8 waves: 2 (M) x 4 (N); wave tile 64x32.
// ---------------------------------------------------------------------------
template <bool A_BF16, bool OUT_BF16>
__global__ __launch_bounds__(256)
void gemm_bias_wmma_kernel(const void* __restrict__ Aptr,
                           const float* __restrict__ W,
                           const float* __restrict__ bias,
                           void* __restrict__ Out,
                           int K, int N) {
  __shared__ bf16_t Asm[128 * 40];   // [m][k], stride 40 (16B aligned rows)
  __shared__ bf16_t Bsm[128 * 40];   // [n][k] (transposed), stride 40

  const int tid  = threadIdx.x;
  const int lane = tid & 31;
  const int wave = tid >> 5;
  const int m0 = blockIdx.y * 128;
  const int n0 = blockIdx.x * 128;
  const int wm = (wave & 1) * 64;   // wave M offset
  const int wn = (wave >> 1) * 32;  // wave N offset

  v8f zero;
  #pragma unroll
  for (int i = 0; i < 8; ++i) zero[i] = 0.0f;
  v8f acc[4][2];
  #pragma unroll
  for (int mi = 0; mi < 4; ++mi)
    #pragma unroll
    for (int ni = 0; ni < 2; ++ni) acc[mi][ni] = zero;

  const int kTiles = K >> 5;
  for (int kt = 0; kt < kTiles; ++kt) {
    __syncthreads();
    // ---- stage A tile 128x32 ----
    #pragma unroll
    for (int p = 0; p < 4; ++p) {
      const int row = p * 32 + (tid >> 3);
      const int c4  = (tid & 7) * 4;
      if (A_BF16) {
        const bf16_t* ap = (const bf16_t*)Aptr + (size_t)(m0 + row) * K + kt * 32 + c4;
        *(v4bf*)&Asm[row * 40 + c4] = *(const v4bf*)ap;
        if (kt + 1 < kTiles) __builtin_prefetch(ap + 32, 0, 1);
      } else {
        const float* ap = (const float*)Aptr + (size_t)(m0 + row) * K + kt * 32 + c4;
        v4f f = *(const v4f*)ap;
        v4bf d;
        #pragma unroll
        for (int i = 0; i < 4; ++i) d[i] = f2bf(f[i]);
        *(v4bf*)&Asm[row * 40 + c4] = d;
        if (kt + 1 < kTiles) __builtin_prefetch(ap + 32, 0, 1);
      }
    }
    // ---- stage W tile 32x128 transposed -> Bsm[n][k] ----
    #pragma unroll
    for (int p = 0; p < 4; ++p) {
      const int kr = p * 8 + (tid >> 5);
      const int c4 = (tid & 31) * 4;
      const float* wp = W + (size_t)(kt * 32 + kr) * N + n0 + c4;
      v4f f = *(const v4f*)wp;
      #pragma unroll
      for (int i = 0; i < 4; ++i) Bsm[(c4 + i) * 40 + kr] = f2bf(f[i]);
      if (kt + 1 < kTiles) __builtin_prefetch(wp + (size_t)32 * N, 0, 1);
    }
    __syncthreads();
    // ---- compute ----
    v16bf af[4], bfr[2];
    #pragma unroll
    for (int mi = 0; mi < 4; ++mi)
      af[mi] = frag_a(&Asm[(wm + mi * 16) * 40], 40, lane);
    #pragma unroll
    for (int ni = 0; ni < 2; ++ni)
      bfr[ni] = frag_b(&Bsm[(wn + ni * 16) * 40], 40, lane);
    #pragma unroll
    for (int mi = 0; mi < 4; ++mi)
      #pragma unroll
      for (int ni = 0; ni < 2; ++ni)
        acc[mi][ni] = wmma_bf16(af[mi], bfr[ni], acc[mi][ni]);
  }

  // ---- epilogue: bias + store (C layout: row = j + 8*half, col = lane%16) ----
  const int half = lane >> 4;
  const int nl   = lane & 15;
  #pragma unroll
  for (int mi = 0; mi < 4; ++mi) {
    #pragma unroll
    for (int ni = 0; ni < 2; ++ni) {
      const int col = n0 + wn + ni * 16 + nl;
      const float bv = bias[col];
      #pragma unroll
      for (int j = 0; j < 8; ++j) {
        const int row = m0 + wm + mi * 16 + j + 8 * half;
        const float v = acc[mi][ni][j] + bv;
        if (OUT_BF16)
          ((bf16_t*)Out)[(size_t)row * N + col] = f2bf(v);
        else
          ((float*)Out)[(size_t)row * N + col] = v;
      }
    }
  }
}

// ---------------------------------------------------------------------------
// Flash MQA attention. Block: 256 threads = 8 waves; one head per wave,
// 16 query rows per block; K/V LDS tiles (32 keys) shared by all 8 heads.
// K tile staged by the Tensor Data Mover; V staged manually (transposed).
// grid = (H/8, S/16, B)
// ---------------------------------------------------------------------------
__global__ __launch_bounds__(256)
void mqa_flash_wmma_kernel(const bf16_t* __restrict__ qkv,
                           bf16_t* __restrict__ aout) {
  __shared__ bf16_t Ksm[32 * KSTR];       // [key][hd]  stride 136 (TDM-padded)
  __shared__ bf16_t Vsm[128 * 40];        // [hd][key]  stride 40 (transposed)
  __shared__ bf16_t Psm[8 * 16 * 40];     // per-wave P scratch [row][key]

  const int tid  = threadIdx.x;
  const int lane = tid & 31;
  const int wave = tid >> 5;
  const int half = lane >> 4;
  const int nl   = lane & 15;

  const int b    = blockIdx.z;
  const int q0   = blockIdx.y * 16;
  const int head = blockIdx.x * 8 + wave;
  const size_t rowbase = (size_t)b * SEQ;
  const unsigned ksm_off = (unsigned)(unsigned long long)(void*)&Ksm[0];

  // Q fragments for this wave's head: 16 rows x 128 (4 chunks of K=32)
  v16bf qf[4];
  {
    const bf16_t* qbase = qkv + (rowbase + q0) * QKVW + head * HDIM;
    #pragma unroll
    for (int kk = 0; kk < 4; ++kk)
      qf[kk] = frag_a(qbase + kk * 32, QKVW, lane);
  }

  v8f zero;
  #pragma unroll
  for (int i = 0; i < 8; ++i) zero[i] = 0.0f;
  v8f o[8];
  #pragma unroll
  for (int nt = 0; nt < 8; ++nt) o[nt] = zero;
  float mrow[8], lrow[8];
  #pragma unroll
  for (int j = 0; j < 8; ++j) { mrow[j] = -3.0e38f; lrow[j] = 0.0f; }

  const float scale = 0.08838834764831845f;  // 1/sqrt(128)
  const int nkt = q0 / 32 + 1;               // causal tile bound

  for (int kt = 0; kt < nkt; ++kt) {
    __syncthreads();  // previous-tile consumers done with Ksm/Vsm
    // ---- stage K tile 32x128 via Tensor Data Mover ----
    if (wave == 0)
      tdm_load_k_tile(qkv + (rowbase + kt * 32) * QKVW + KCOL, ksm_off);
    // ---- stage V tile 32x128, transposed -> Vsm[hd][key] (all threads) ----
    #pragma unroll
    for (int p = 0; p < 2; ++p) {
      const int idx = tid + p * 256;
      const int r  = idx >> 4;
      const int hc = (idx & 15) * 8;
      v8bf d = *(const v8bf*)(qkv + (rowbase + kt * 32 + r) * QKVW + VCOL + hc);
      #pragma unroll
      for (int i = 0; i < 8; ++i) Vsm[(hc + i) * 40 + r] = d[i];
    }
    if (wave == 0)
      __builtin_amdgcn_s_wait_tensorcnt(0);  // TDM complete before release
    __syncthreads();

    // ---- scores S = Q @ K^T : two 16x16 tiles (keys 0-15, 16-31) ----
    v8f s0 = zero, s1 = zero;
    #pragma unroll
    for (int kk = 0; kk < 4; ++kk) {
      v16bf kf0 = frag_b(&Ksm[0 * KSTR + kk * 32], KSTR, lane);
      v16bf kf1 = frag_b(&Ksm[16 * KSTR + kk * 32], KSTR, lane);
      s0 = wmma_bf16(qf[kk], kf0, s0);
      s1 = wmma_bf16(qf[kk], kf1, s1);
    }

    // ---- online softmax (per row j+8*half, reduce across 16-lane group) ----
    bf16_t* Pb = &Psm[wave * 16 * 40];
    #pragma unroll
    for (int j = 0; j < 8; ++j) {
      const int qg  = q0 + j + 8 * half;
      const int kg0 = kt * 32 + nl;
      float v0 = s0[j] * scale;  if (kg0 > qg)      v0 = -3.0e38f;
      float v1 = s1[j] * scale;  if (kg0 + 16 > qg) v1 = -3.0e38f;
      float pm = fmaxf(v0, v1);
      #pragma unroll
      for (int off = 8; off > 0; off >>= 1)
        pm = fmaxf(pm, __shfl_xor(pm, off, 16));
      const float nm    = fmaxf(mrow[j], pm);
      const float alpha = __expf(mrow[j] - nm);
      mrow[j] = nm;
      const float p0 = __expf(v0 - nm);
      const float p1 = __expf(v1 - nm);
      float ps = p0 + p1;
      #pragma unroll
      for (int off = 8; off > 0; off >>= 1)
        ps += __shfl_xor(ps, off, 16);
      lrow[j] = lrow[j] * alpha + ps;
      #pragma unroll
      for (int nt = 0; nt < 8; ++nt) o[nt][j] *= alpha;
      // C-layout -> A-layout transpose through LDS
      Pb[(j + 8 * half) * 40 + nl]      = f2bf(p0);
      Pb[(j + 8 * half) * 40 + 16 + nl] = f2bf(p1);
    }
    asm volatile("s_wait_dscnt 0" ::: "memory");  // P writes -> P reads (same wave)

    // ---- O += P @ V ----
    v16bf pf = frag_a(Pb, 40, lane);
    #pragma unroll
    for (int nt = 0; nt < 8; ++nt) {
      v16bf vf = frag_b(&Vsm[nt * 16 * 40], 40, lane);
      o[nt] = wmma_bf16(pf, vf, o[nt]);
    }
  }

  // ---- normalize + store attn output (bf16, [tok][head*128 + hd]) ----
  #pragma unroll
  for (int nt = 0; nt < 8; ++nt) {
    #pragma unroll
    for (int j = 0; j < 8; ++j) {
      const int row = q0 + j + 8 * half;
      const int col = head * HDIM + nt * 16 + nl;
      const float v = o[nt][j] / lrow[j];
      aout[(rowbase + row) * DMODEL + col] = f2bf(v);
    }
  }
}

// ---------------------------------------------------------------------------
extern "C" void kernel_launch(void* const* d_in, const int* in_sizes, int n_in,
                              void* d_out, int out_size, void* d_ws, size_t ws_size,
                              hipStream_t stream) {
  (void)in_sizes; (void)n_in; (void)out_size; (void)ws_size;
  const float* x      = (const float*)d_in[0];  // [B,S,D]
  const float* w_attn = (const float*)d_in[1];  // [D, QKVW]
  const float* b_attn = (const float*)d_in[2];  // [QKVW]
  const float* w_proj = (const float*)d_in[3];  // [D, D]
  const float* b_proj = (const float*)d_in[4];  // [D]
  float* out = (float*)d_out;                   // [B,S,D]

  const int M = BATCH * SEQ;                         // 4096
  bf16_t* qkv_ws  = (bf16_t*)d_ws;                   // M x QKVW bf16
  bf16_t* attn_ws = (bf16_t*)((char*)d_ws + (size_t)M * QKVW * sizeof(bf16_t));

  // 1) qkv = x @ w_attn + b_attn  (f32 -> bf16)
  gemm_bias_wmma_kernel<false, true>
      <<<dim3(QKVW / 128, M / 128), 256, 0, stream>>>(
          (const void*)x, w_attn, b_attn, (void*)qkv_ws, DMODEL, QKVW);

  // 2) flash MQA attention (TDM-staged K tiles)
  mqa_flash_wmma_kernel
      <<<dim3(NHEAD / 8, SEQ / 16, BATCH), 256, 0, stream>>>(qkv_ws, attn_ws);

  // 3) out = attn @ w_proj + b_proj  (bf16 -> f32)
  gemm_bias_wmma_kernel<true, false>
      <<<dim3(DMODEL / 128, M / 128), 256, 0, stream>>>(
          (const void*)attn_ws, w_proj, b_proj, (void*)out, DMODEL, DMODEL);
}
